// SwinTransformerBlock_3066606649868
// MI455X (gfx1250) — matrix operand order repaired and checked by reading, not compiled, not run on hardware
//
#include <hip/hip_runtime.h>
#include <hip/hip_bf16.h>

typedef __bf16 bf16;
typedef __attribute__((ext_vector_type(16))) __bf16 bf16x16;
typedef __attribute__((ext_vector_type(8)))  float  f32x8;

union Frag { bf16x16 v; uint4 q[2]; };

constexpr int ROWS = 50176;   // 16 images * 64 windows * 49 tokens

__device__ __forceinline__ int region_of(int p) { return p < 49 ? 0 : (p < 53 ? 1 : 2); }

// ---------------------------------------------------------------------------
// LayerNorm (+ optional shift/window-partition gather), fp32 in -> bf16 out.
// One wave (32 lanes) per 512-channel row; lane owns 16 contiguous channels.
// ---------------------------------------------------------------------------
__global__ __launch_bounds__(256) void ln_kernel(const float* __restrict__ x,
                                                 const float* __restrict__ g,
                                                 const float* __restrict__ b,
                                                 bf16* __restrict__ out,
                                                 int shifted) {
  int wave = threadIdx.x >> 5, lane = threadIdx.x & 31;
  int r = blockIdx.x * 8 + wave;            // output row (window order if shifted)
  int src;
  if (shifted) {
    int w = r / 49, n = r % 49;
    int bb = w >> 6, wi = w & 63;
    int wh = wi >> 3, ww = wi & 7;
    int i = n / 7, j = n % 7;
    int sh = (wh * 7 + i + 3) % 56;         // roll(-3): xs[h] = xn[(h+3)%56]
    int sw = (ww * 7 + j + 3) % 56;
    src = bb * 3136 + sh * 56 + sw;
  } else {
    src = r;
  }
  const float* xp = x + (size_t)src * 512 + lane * 16;
  float v[16];
  for (int t = 0; t < 4; ++t) {
    float4 f = *(const float4*)(xp + t * 4);
    v[t*4+0] = f.x; v[t*4+1] = f.y; v[t*4+2] = f.z; v[t*4+3] = f.w;
  }
  float s1 = 0.f, s2 = 0.f;
  for (int i = 0; i < 16; ++i) { s1 += v[i]; s2 += v[i] * v[i]; }
  for (int off = 16; off; off >>= 1) {
    s1 += __shfl_xor(s1, off, 32);
    s2 += __shfl_xor(s2, off, 32);
  }
  float mu  = s1 * (1.f / 512.f);
  float var = s2 * (1.f / 512.f) - mu * mu;
  float rs  = rsqrtf(var + 1e-5f);
  int c0 = lane * 16;
  bf16* op = out + (size_t)r * 512 + c0;
  for (int i = 0; i < 16; ++i) op[i] = (bf16)((v[i] - mu) * rs * g[c0 + i] + b[c0 + i]);
}

// ---------------------------------------------------------------------------
// Tiled bf16 WMMA GEMM: C[M,N] = A[M,K](bf16) * B[K,N](fp32 weights, converted
// on LDS fill) with mode-specific epilogues.
//   MODE 0: QKV  -> +bias, q-cols scaled, bf16 out [M,N]
//   MODE 1: PROJ -> +bias, window-reverse/unshift, +x residual, fp32 y
//   MODE 2: FC1  -> +bias, exact GELU, bf16 out
//   MODE 3: FC2  -> +bias, +y residual, fp32 d_out
// Block: 128x128 tile, BK=32, 8 waves (2x4), wave tile 64x32 -> 4x2 WMMA accs.
// ---------------------------------------------------------------------------
template<int MODE>
__global__ __launch_bounds__(256) void gemm_kernel(const bf16*  __restrict__ A,
                                                   const float* __restrict__ Bw,
                                                   const float* __restrict__ bias,
                                                   const float* __restrict__ res,
                                                   void* __restrict__ outv,
                                                   int N, int K) {
  __shared__ __align__(16) bf16 As[128][40];   // row-major, pad 32->40 halfwords
  __shared__ __align__(16) bf16 Bs[128][40];   // B transposed: [n][k]
  int tid  = threadIdx.x;
  int wave = tid >> 5, lane = tid & 31;
  int wm = wave >> 2, wn = wave & 3;           // 2x4 wave grid
  int lh = lane >> 4, lr = lane & 15;
  int m0 = blockIdx.y * 128, n0 = blockIdx.x * 128;

  f32x8 acc[4][2];
  for (int i = 0; i < 4; ++i)
    for (int j = 0; j < 2; ++j)
      acc[i][j] = f32x8{0.f,0.f,0.f,0.f,0.f,0.f,0.f,0.f};

  for (int k0 = 0; k0 < K; k0 += 32) {
    // Prefetch next K tile into cache (global_prefetch_b8) while this one runs
    if (k0 + 32 < K) {
      __builtin_prefetch((const void*)(A + (size_t)(m0 + (tid & 127)) * K + k0 + 32), 0, 3);
      if (tid < 32)
        __builtin_prefetch((const void*)(Bw + (size_t)(k0 + 32 + tid) * N + n0), 0, 3);
    }
    // A tile: 128x32 bf16; 512 chunks of 16B, 2 per thread
    for (int it = 0; it < 2; ++it) {
      int ch = tid + it * 256;
      int row = ch >> 2, seg = ch & 3;
      *(uint4*)&As[row][seg * 8] =
          *(const uint4*)(A + (size_t)(m0 + row) * K + k0 + seg * 8);
    }
    // B tile: 32x128 fp32 -> bf16 transposed into Bs[n][k]
    for (int it = 0; it < 4; ++it) {
      int ch = tid + it * 256;
      int row = ch >> 5, c0 = (ch & 31) * 4;
      float4 f = *(const float4*)(Bw + (size_t)(k0 + row) * N + n0 + c0);
      Bs[c0 + 0][row] = (bf16)f.x;
      Bs[c0 + 1][row] = (bf16)f.y;
      Bs[c0 + 2][row] = (bf16)f.z;
      Bs[c0 + 3][row] = (bf16)f.w;
    }
    __syncthreads();
    // B fragments: col = wn*32 + tj*16 + lr, K = (i/8)*16 + lh*8 + i%8
    Frag bfr[2];
    for (int tj = 0; tj < 2; ++tj) {
      const bf16* bp = &Bs[wn * 32 + tj * 16 + lr][lh * 8];
      bfr[tj].q[0] = *(const uint4*)bp;
      bfr[tj].q[1] = *(const uint4*)(bp + 16);
    }
    // A fragments: row = wm*64 + ti*16 + lr, K = (i/8)*16 + lh*8 + i%8
    for (int ti = 0; ti < 4; ++ti) {
      Frag af;
      const bf16* ap = &As[wm * 64 + ti * 16 + lr][lh * 8];
      af.q[0] = *(const uint4*)ap;
      af.q[1] = *(const uint4*)(ap + 16);
      for (int tj = 0; tj < 2; ++tj)
        acc[ti][tj] = __builtin_amdgcn_wmma_f32_16x16x32_bf16(
            false, af.v, false, bfr[tj].v, (short)0, acc[ti][tj], false, false);
    }
    __syncthreads();
  }

  // Epilogue. C layout: row = vg + 8*lh (within 16x16 tile), col = lr.
  for (int ti = 0; ti < 4; ++ti) {
    for (int vg = 0; vg < 8; ++vg) {
      int gm = m0 + wm * 64 + ti * 16 + vg + 8 * lh;
      size_t orow = 0;
      if (MODE == 1) {                         // window reverse + unshift
        int w = gm / 49, n = gm % 49;
        int bb = w >> 6, wi = w & 63;
        int wh = wi >> 3, ww = wi & 7;
        int i = n / 7, j = n % 7;
        int dh = (wh * 7 + i + 3) % 56;        // roll(+3): out[(hs+3)%56] = xs[hs]
        int dw = (ww * 7 + j + 3) % 56;
        orow = (size_t)(bb * 3136 + dh * 56 + dw);
      }
      for (int tj = 0; tj < 2; ++tj) {
        int gn = n0 + wn * 32 + tj * 16 + lr;
        float val = acc[ti][tj][vg] + bias[gn];
        if (MODE == 0) {
          if (gn < 512) val *= 0.17677669529663689f;    // 1/sqrt(hd=32) on q
          ((bf16*)outv)[(size_t)gm * N + gn] = (bf16)val;
        } else if (MODE == 1) {
          size_t idx = orow * 512 + gn;
          ((float*)outv)[idx] = res[idx] + val;
        } else if (MODE == 2) {
          float ge = 0.5f * val * (1.0f + erff(val * 0.7071067811865475f));
          ((bf16*)outv)[(size_t)gm * N + gn] = (bf16)ge;
        } else {
          size_t idx = (size_t)gm * 512 + gn;
          ((float*)outv)[idx] = res[idx] + val;
        }
      }
    }
  }
}

// ---------------------------------------------------------------------------
// Windowed attention: one wave per (window, head). q pre-scaled in QKV GEMM.
// S = q k^T (16 WMMAs); bias+mask folded into the fp32 softmax pass (rpb head
// slice + region codes preloaded to LDS -- keeps the WMMA loop free of VALU
// index math and global loads); out = P v (16 WMMAs). Rows/cols 49..63 pad.
// ---------------------------------------------------------------------------
__global__ __launch_bounds__(32) void attn_kernel(const bf16*  __restrict__ qkv,
                                                  const float* __restrict__ rpb,
                                                  bf16* __restrict__ aout) {
  __shared__ __align__(16) bf16 qs[64][40];
  __shared__ __align__(16) bf16 ks[64][40];
  __shared__ __align__(16) bf16 vs[64][40];
  __shared__ __align__(16) float Ss[64][68];
  __shared__ __align__(16) bf16  Ps[64][72];
  __shared__ float rb[169];     // rpb_table[:, head]
  __shared__ int   rg[64];      // shift-mask region code per token

  int lane = threadIdx.x;
  int w  = blockIdx.x >> 4;        // global window 0..1023
  int hh = blockIdx.x & 15;        // head
  int wi = w & 63;                 // window within image (for the mask)
  int wh = wi >> 3, ww = wi & 7;
  int lh = lane >> 4, lr = lane & 15;

  // Preload per-head rel-pos bias and per-token region codes.
  for (int t = lane; t < 169; t += 32) rb[t] = rpb[t * 16 + hh];
  for (int t = lane; t < 64; t += 32) {
    int i = t / 7, j = t % 7;
    rg[t] = (t < 49) ? region_of(wh * 7 + i) * 3 + region_of(ww * 7 + j) : 0;
  }

  for (int n = lane; n < 64; n += 32) {
    if (n < 49) {
      const bf16* src = qkv + (size_t)(w * 49 + n) * 1536 + hh * 32;
      for (int t = 0; t < 4; ++t) {
        *(uint4*)&qs[n][t * 8] = *(const uint4*)(src + t * 8);
        *(uint4*)&ks[n][t * 8] = *(const uint4*)(src + 512 + t * 8);
        *(uint4*)&vs[n][t * 8] = *(const uint4*)(src + 1024 + t * 8);
      }
    } else {
      uint4 z{0, 0, 0, 0};
      for (int t = 0; t < 4; ++t) {
        *(uint4*)&qs[n][t * 8] = z;
        *(uint4*)&ks[n][t * 8] = z;
        *(uint4*)&vs[n][t * 8] = z;
      }
    }
  }
  __syncthreads();

  // S = q k^T  (K = head dim = 32, one WMMA per 16x16 tile); raw scores only.
  for (int ti = 0; ti < 4; ++ti) {
    Frag af;
    const bf16* ap = &qs[ti * 16 + lr][lh * 8];
    af.q[0] = *(const uint4*)ap;
    af.q[1] = *(const uint4*)(ap + 16);
    for (int tj = 0; tj < 4; ++tj) {
      Frag bfr;                                   // B[k][m] = k[m][k]
      const bf16* bp = &ks[tj * 16 + lr][lh * 8];
      bfr.q[0] = *(const uint4*)bp;
      bfr.q[1] = *(const uint4*)(bp + 16);
      f32x8 c = f32x8{0.f,0.f,0.f,0.f,0.f,0.f,0.f,0.f};
      c = __builtin_amdgcn_wmma_f32_16x16x32_bf16(
          false, af.v, false, bfr.v, (short)0, c, false, false);
      float* srow = &Ss[ti * 16 + 8 * lh][tj * 16 + lr];
      for (int vg = 0; vg < 8; ++vg) srow[vg * 68] = c[vg];
    }
  }
  __syncthreads();

  // Softmax with fused bias+mask; P in bf16, padded cols zeroed.
  for (int r = lane; r < 64; r += 32) {
    if (r < 49) {
      int i1 = r / 7, j1 = r % 7;
      int base = (i1 + 6) * 13 + (j1 + 6);        // rel = base - (i2*13 + j2)
      int rgr = rg[r];
      float mx = -1e30f;
      int m = 0;
      for (int i2 = 0; i2 < 7; ++i2) {
        int bi = base - i2 * 13;
        for (int j2 = 0; j2 < 7; ++j2, ++m) {
          float val = Ss[r][m] + rb[bi - j2] + ((rgr == rg[m]) ? 0.f : -100.f);
          Ss[r][m] = val;
          mx = fmaxf(mx, val);
        }
      }
      float s = 0.f;
      for (int mm = 0; mm < 49; ++mm) { float e = __expf(Ss[r][mm] - mx); Ss[r][mm] = e; s += e; }
      float inv = 1.f / s;
      for (int mm = 0; mm < 49; ++mm) Ps[r][mm] = (bf16)(Ss[r][mm] * inv);
      for (int mm = 49; mm < 64; ++mm) Ps[r][mm] = (bf16)0.f;
    } else {
      for (int mm = 0; mm < 64; ++mm) Ps[r][mm] = (bf16)0.f;
    }
  }
  __syncthreads();

  // out = P v   (M tiles 4, N tiles 2 over 32 dims, K = 64 keys -> 2 steps)
  for (int ti = 0; ti < 4; ++ti) {
    for (int tn = 0; tn < 2; ++tn) {
      f32x8 c = f32x8{0.f,0.f,0.f,0.f,0.f,0.f,0.f,0.f};
      for (int kspl = 0; kspl < 2; ++kspl) {
        Frag af, bfr;
        const bf16* ap = &Ps[ti * 16 + lr][kspl * 32 + lh * 8];
        af.q[0] = *(const uint4*)ap;
        af.q[1] = *(const uint4*)(ap + 16);
        for (int i = 0; i < 16; ++i)             // B[k][d] = v[key][d]
          bfr.v[i] = vs[kspl * 32 + (i >> 3) * 16 + lh * 8 + (i & 7)][tn * 16 + lr];
        c = __builtin_amdgcn_wmma_f32_16x16x32_bf16(
            false, af.v, false, bfr.v, (short)0, c, false, false);
      }
      for (int vg = 0; vg < 8; ++vg) {
        int n = ti * 16 + vg + 8 * lh;
        if (n < 49)
          aout[(size_t)(w * 49 + n) * 512 + hh * 32 + tn * 16 + lr] = (bf16)c[vg];
      }
    }
  }
}

// ---------------------------------------------------------------------------
extern "C" void kernel_launch(void* const* d_in, const int* in_sizes, int n_in,
                              void* d_out, int out_size, void* d_ws, size_t ws_size,
                              hipStream_t stream) {
  const float* x      = (const float*)d_in[0];
  const float* n1g    = (const float*)d_in[1];
  const float* n1b    = (const float*)d_in[2];
  const float* qkv_w  = (const float*)d_in[3];
  const float* qkv_b  = (const float*)d_in[4];
  const float* rpb    = (const float*)d_in[5];
  const float* proj_w = (const float*)d_in[6];
  const float* proj_b = (const float*)d_in[7];
  const float* n2g    = (const float*)d_in[8];
  const float* n2b    = (const float*)d_in[9];
  const float* fc1_w  = (const float*)d_in[10];
  const float* fc1_b  = (const float*)d_in[11];
  const float* fc2_w  = (const float*)d_in[12];
  const float* fc2_b  = (const float*)d_in[13];

  // Workspace arena with lifetime-based overlay (peak ~343 MB):
  //   [0, SZ_Y)            : A_ln (bf16, K0..K1)  then  y (fp32, K3..K6)
  //   [SZ_Y, SZ_Y+SZ_H)    : attn_out (bf16, K2..K3) then h1 (bf16, K4..K5)
  //   [SZ_Y+SZ_H, ...)     : qkv (bf16, K1..K2) then h2 (bf16, K5..K6)
  char* ws = (char*)d_ws;
  const size_t SZ_Y = (size_t)ROWS * 512 * 4;
  const size_t SZ_H = (size_t)ROWS * 512 * 2;
  bf16*  A_ln = (bf16*)(ws);
  float* y    = (float*)(ws);
  bf16*  attn = (bf16*)(ws + SZ_Y);
  bf16*  h1   = (bf16*)(ws + SZ_Y);
  bf16*  qkvb = (bf16*)(ws + SZ_Y + SZ_H);
  bf16*  h2   = (bf16*)(ws + SZ_Y + SZ_H);

  // K0: LN1 + shift + window partition -> bf16 A [50176, 512]
  ln_kernel<<<ROWS / 8, 256, 0, stream>>>(x, n1g, n1b, A_ln, 1);
  // K1: QKV GEMM [50176,512]x[512,1536]
  gemm_kernel<0><<<dim3(12, 392), 256, 0, stream>>>(A_ln, qkv_w, qkv_b, nullptr,
                                                    (void*)qkvb, 1536, 512);
  // K2: windowed attention (1024 windows x 16 heads)
  attn_kernel<<<1024 * 16, 32, 0, stream>>>(qkvb, rpb, attn);
  // K3: proj GEMM + window reverse + unshift + residual(x) -> y fp32
  gemm_kernel<1><<<dim3(4, 392), 256, 0, stream>>>(attn, proj_w, proj_b, x,
                                                   (void*)y, 512, 512);
  // K4: LN2 -> bf16
  ln_kernel<<<ROWS / 8, 256, 0, stream>>>(y, n2g, n2b, h1, 0);
  // K5: FC1 GEMM + GELU -> bf16 [50176, 2048]
  gemm_kernel<2><<<dim3(16, 392), 256, 0, stream>>>(h1, fc1_w, fc1_b, nullptr,
                                                    (void*)h2, 2048, 512);
  // K6: FC2 GEMM + residual(y) -> d_out fp32
  gemm_kernel<3><<<dim3(4, 392), 256, 0, stream>>>(h2, fc2_w, fc2_b, y,
                                                   d_out, 512, 2048);
}
